// GPTLayer_49830210568467
// MI455X (gfx1250) — matrix operand order repaired and checked by reading, not compiled, not run on hardware
//
#include <hip/hip_runtime.h>

// ---------------------------------------------------------------------------
// GPT block for MI455X (gfx1250, wave32).
// All GEMMs + attention einsums on v_wmma_f32_16x16x32_f16 (f16 in, f32 acc).
// GEMM stages tiles with global_load_async_to_lds_b128 (ASYNCcnt path, no
// VGPR round-trip); softmax row reductions use DPP row_ror rotations (VALU,
// co-executes with WMMA); softmax runs in the log2 domain (v_exp_f32 is
// base-2 native) with log2(e)/sqrt(hd) folded into Q.
// B=2 T=2048 C=1024 H=16 hd=64 F=4096, M = B*T = 4096.
// ---------------------------------------------------------------------------

typedef _Float16 half_t;
typedef __attribute__((ext_vector_type(16))) _Float16 v16h;
typedef __attribute__((ext_vector_type(8)))  _Float16 v8h;
typedef __attribute__((ext_vector_type(4)))  _Float16 v4h;
typedef __attribute__((ext_vector_type(8)))  float    v8f;
typedef __attribute__((ext_vector_type(4)))  float    v4f;

#define DEVFN static __device__ __forceinline__

DEVFN v16h combine(v8h lo, v8h hi) {
  v16h r;
#pragma unroll
  for (int i = 0; i < 8; ++i) { r[i] = lo[i]; r[i + 8] = hi[i]; }
  return r;
}

DEVFN v8f wmma(v16h a, v16h b, v8f c) {
  // D = A(16x32 f16) * B(32x16 f16) + C(16x16 f32)
  return __builtin_amdgcn_wmma_f32_16x16x32_f16(false, a, false, b, (short)0, c,
                                                false, false);
}

// --- DPP 16-lane row reductions (row_ror:N = 0x120+N), pure VALU ------------
template <int CTRL>
DEVFN float dpp_rot(float x) {
  return __builtin_bit_cast(
      float, __builtin_amdgcn_update_dpp(0, __builtin_bit_cast(int, x), CTRL,
                                         0xf, 0xf, true));
}
DEVFN float row_sum16(float x) {
  x += dpp_rot<0x121>(x);  // ror:1
  x += dpp_rot<0x122>(x);  // ror:2
  x += dpp_rot<0x124>(x);  // ror:4
  x += dpp_rot<0x128>(x);  // ror:8
  return x;
}
DEVFN float row_max16(float x) {
  x = fmaxf(x, dpp_rot<0x121>(x));
  x = fmaxf(x, dpp_rot<0x122>(x));
  x = fmaxf(x, dpp_rot<0x124>(x));
  x = fmaxf(x, dpp_rot<0x128>(x));
  return x;
}

// --- async copy global -> LDS (CDNA5 ASYNCcnt path) -------------------------
DEVFN unsigned lds_off(const void* p) { return (unsigned)(size_t)p; }
DEVFN void async_b128(unsigned lds, const half_t* g) {
  asm volatile("global_load_async_to_lds_b128 %0, %1, off"
               :: "v"(lds), "v"(g) : "memory");
}
DEVFN void wait_async0() { asm volatile("s_wait_asynccnt 0" ::: "memory"); }

// ---------------------------------------------------------------------------
// Weight transpose + f32->f16 convert:  dst[n*K + k] = (f16) src[k*N + n]
// ---------------------------------------------------------------------------
__global__ __launch_bounds__(256) void convT(const float* __restrict__ src,
                                             half_t* __restrict__ dst,
                                             int K, int N) {
  size_t total = (size_t)K * N;
  size_t stride = (size_t)gridDim.x * 256u;
  for (size_t i = (size_t)blockIdx.x * 256u + threadIdx.x; i < total; i += stride) {
    size_t n = i / (size_t)K;
    size_t k = i - n * (size_t)K;
    dst[i] = (half_t)src[k * (size_t)N + n];
  }
}

// ---------------------------------------------------------------------------
// Fused residual scale + RMSNorm over C=1024. One block per token row.
// ---------------------------------------------------------------------------
__global__ __launch_bounds__(256) void resid_rmsnorm(
    const float* __restrict__ x, const float* __restrict__ x0,
    const float* __restrict__ pLa, const float* __restrict__ pLb,
    float* __restrict__ scaledOut, half_t* __restrict__ norm16, int doResid) {
  const int tid = threadIdx.x;
  const size_t base = (size_t)blockIdx.x * 1024 + tid * 4;
  v4f v = *(const v4f*)(x + base);
  if (doResid) {
    const float la = pLa[0], lb = pLb[0];
    v4f w = *(const v4f*)(x0 + base);
#pragma unroll
    for (int i = 0; i < 4; ++i) v[i] = v[i] * la + w[i] * lb;
    *(v4f*)(scaledOut + base) = v;
  }
  float ss = v[0] * v[0] + v[1] * v[1] + v[2] * v[2] + v[3] * v[3];
  ss = row_sum16(ss);
  ss += __shfl_xor(ss, 16, 32);
  __shared__ float red[8];
  __shared__ float rtot;
  if ((tid & 31) == 0) red[tid >> 5] = ss;
  __syncthreads();
  if (tid == 0) {
    float s = 0.f;
#pragma unroll
    for (int i = 0; i < 8; ++i) s += red[i];
    rtot = rsqrtf(s * (1.0f / 1024.0f) + 1e-5f);
  }
  __syncthreads();
  const float rms = rtot;
  v4h o;
#pragma unroll
  for (int i = 0; i < 4; ++i) o[i] = (half_t)(v[i] * rms);
  *(v4h*)(norm16 + base) = o;
}

// ---------------------------------------------------------------------------
// WMMA GEMM: Out[M][N] = A[M][K](f16) * Bt[N][K](f16, pre-transposed)
// Block tile 128(M) x 256(N), K-step 32. 8 waves (2x4), wave tile 64x64:
// 16 WMMAs per K-step fed by 16 ds_read_b128 (1.0 LDS read / WMMA).
// Double-buffered LDS filled by global_load_async_to_lds_b128 (6 per thread
// per step); padded row stride (40 halves) spreads banks.
// MODE 0: f16 out.  MODE 1: f32 out = Resid + acc.
// MODE 2: f16 out = relu(acc)^2.  MODE 3: f16 out scattered to [b][h][d][t].
// ---------------------------------------------------------------------------
template <int MODE>
__global__ __launch_bounds__(256) void gemm256(
    const half_t* __restrict__ A, const half_t* __restrict__ Bt,
    void* __restrict__ OutP, const float* __restrict__ Resid,
    int Mdim, int Ndim, int Kdim) {
  constexpr int LDSS = 40;  // padded row stride in halves (80B)
  __shared__ half_t sA[2][128 * LDSS];
  __shared__ half_t sB[2][256 * LDSS];

  const int tid = threadIdx.x;
  const int lane = tid & 31, wid = tid >> 5;
  const int lane16 = lane & 15, sel = lane >> 4;
  const int wm = wid >> 2, wn = wid & 3;  // 2 x 4 wave grid
  const int blockM = blockIdx.y * 128, blockN = blockIdx.x * 256;

  // Staging: thread owns one 32B segment of one A row and of two B rows.
  const int lrow = tid >> 1;          // 0..127
  const int lseg = (tid & 1) * 16;    // 0 or 16 halves
  const half_t* gA  = A  + (size_t)(blockM + lrow) * Kdim + lseg;
  const half_t* gB0 = Bt + (size_t)(blockN + lrow) * Kdim + lseg;
  const half_t* gB1 = Bt + (size_t)(blockN + 128 + lrow) * Kdim + lseg;
  unsigned la[2], lb0[2], lb1[2];
#pragma unroll
  for (int s = 0; s < 2; ++s) {
    la[s]  = lds_off(&sA[s][lrow * LDSS + lseg]);
    lb0[s] = lds_off(&sB[s][lrow * LDSS + lseg]);
    lb1[s] = lds_off(&sB[s][(lrow + 128) * LDSS + lseg]);
  }

  v8f acc[4][4];
  const v8f vzero = {};
#pragma unroll
  for (int i = 0; i < 4; ++i)
#pragma unroll
    for (int j = 0; j < 4; ++j) acc[i][j] = vzero;

  const int KT = Kdim >> 5;
  int cur = 0;

  auto stage = [&](int buf, int kofs) {
    async_b128(la[buf],       gA  + kofs);
    async_b128(la[buf]  + 16, gA  + kofs + 8);
    async_b128(lb0[buf],      gB0 + kofs);
    async_b128(lb0[buf] + 16, gB0 + kofs + 8);
    async_b128(lb1[buf],      gB1 + kofs);
    async_b128(lb1[buf] + 16, gB1 + kofs + 8);
  };

  // prologue: async-stage tile 0
  stage(0, 0);
  wait_async0();
  __syncthreads();

  for (int kt = 0; kt < KT; ++kt) {
    const bool more = (kt + 1 < KT);
    if (more) stage(cur ^ 1, (kt + 1) * 32);  // DMA overlaps WMMA stream below

    v16h aF[4], bF[4];
#pragma unroll
    for (int i = 0; i < 4; ++i) {
      // A fragment (ISA 16-bit A layout): lane<16 K={0..7,16..23}, else {8..15,24..31}
      const half_t* base = &sA[cur][(wm * 64 + i * 16 + lane16) * LDSS + sel * 8];
      aF[i] = combine(*(const v8h*)base, *(const v8h*)(base + 16));
    }
#pragma unroll
    for (int j = 0; j < 4; ++j) {
      // B fragment (same as A): lane<16 K={0..7,16..23}, else {8..15,24..31}
      const half_t* base = &sB[cur][(wn * 64 + j * 16 + lane16) * LDSS + sel * 8];
      bF[j] = combine(*(const v8h*)base, *(const v8h*)(base + 16));
    }
#pragma unroll
    for (int i = 0; i < 4; ++i)
#pragma unroll
      for (int j = 0; j < 4; ++j) acc[i][j] = wmma(aF[i], bF[j], acc[i][j]);

    if (more) {
      wait_async0();
      __syncthreads();
      cur ^= 1;
    }
  }

  // Epilogue. C-layout: lane -> column n, VGPR r -> row (r + 8*sel).
#pragma unroll
  for (int i = 0; i < 4; ++i)
#pragma unroll
    for (int j = 0; j < 4; ++j) {
      const int n = blockN + wn * 64 + j * 16 + lane16;
#pragma unroll
      for (int r = 0; r < 8; ++r) {
        const int m = blockM + wm * 64 + i * 16 + r + sel * 8;
        const float val = acc[i][j][r];
        if constexpr (MODE == 0) {
          ((half_t*)OutP)[(size_t)m * Ndim + n] = (half_t)val;
        } else if constexpr (MODE == 1) {
          ((float*)OutP)[(size_t)m * Ndim + n] =
              Resid[(size_t)m * Ndim + n] + val;
        } else if constexpr (MODE == 2) {
          const float t = fmaxf(val, 0.f);
          ((half_t*)OutP)[(size_t)m * Ndim + n] = (half_t)(t * t);
        } else {  // V transposed to [b][h][d][t]  (b=m/2048, t=m%2048, h=n/64, d=n%64)
          const int b = m >> 11, t = m & 2047, h = n >> 6, d = n & 63;
          ((half_t*)OutP)[(((size_t)b * 16 + h) * 64 + d) * 2048 + t] = (half_t)val;
        }
      }
    }
}

// ---------------------------------------------------------------------------
// Rotary + per-head RMSNorm (in place), one wave per (b,t,h); lane owns (d, d+32).
// log2(e)/sqrt(hd) folded into Q so attention softmax runs in the log2 domain.
// ---------------------------------------------------------------------------
__global__ __launch_bounds__(256) void rotary_rms(
    half_t* __restrict__ q, half_t* __restrict__ k,
    const float* __restrict__ cosT, const float* __restrict__ sinT) {
  const int widG = blockIdx.x * 8 + (threadIdx.x >> 5);
  const int lane = threadIdx.x & 31;
  const int h = widG & 15, t = (widG >> 4) & 2047, b = widG >> 15;
  const size_t base = ((size_t)(b * 2048 + t)) * 1024 + h * 64;
  const float c1 = cosT[t * 64 + lane], s1 = sinT[t * 64 + lane];
  const float c2 = cosT[t * 64 + 32 + lane], s2 = sinT[t * 64 + 32 + lane];
#pragma unroll
  for (int which = 0; which < 2; ++which) {
    half_t* p = which ? k : q;
    const float scale = which ? 1.0f : 0.125f * 1.44269504088896f;
    const float x1 = (float)p[base + lane], x2 = (float)p[base + 32 + lane];
    const float y1 = x1 * c1 - x2 * s1;   // half_rot first half = -x2
    const float y2 = x2 * c2 + x1 * s2;   // half_rot second half = +x1
    float ss = y1 * y1 + y2 * y2;
    ss = row_sum16(ss);
    ss += __shfl_xor(ss, 16, 32);
    const float rms = rsqrtf(ss * (1.0f / 64.0f) + 1e-5f) * scale;
    p[base + lane] = (half_t)(y1 * rms);
    p[base + 32 + lane] = (half_t)(y2 * rms);
  }
}

// ---------------------------------------------------------------------------
// Flash attention: one wave per (b, h, 16-query tile). 32-key tiles.
// Only the diagonal tile needs the causal mask -> peeled MASKED instantiation
// keeps the steady-state loop select-free. Softmax in log2 domain (v_exp_f32).
// ---------------------------------------------------------------------------
template <bool MASKED>
DEVFN void attn_tile(int j0, int m0, int bT, int hoff, size_t vbase,
                     int lane16, int sel,
                     const half_t* __restrict__ k, const half_t* __restrict__ vT,
                     half_t* myP, v16h a0, v16h a1,
                     float (&mrow)[8], float (&lrow)[8], v8f (&oacc)[4]) {
  constexpr int PS = 40;
  const v8f vzero = {};
  v8f s[2] = {vzero, vzero};
#pragma unroll
  for (int nt = 0; nt < 2; ++nt) {
    const half_t* kp = k + ((size_t)(bT + j0 + nt * 16 + lane16)) * 1024 + hoff +
                       sel * 8;
    const v16h b0 = combine(*(const v8h*)kp, *(const v8h*)(kp + 16));
    const v16h b1 = combine(*(const v8h*)(kp + 32), *(const v8h*)(kp + 48));
    s[nt] = wmma(a0, b0, s[nt]);
    s[nt] = wmma(a1, b1, s[nt]);
  }
#pragma unroll
  for (int r = 0; r < 8; ++r) {
    const int rowAbs = m0 + r + sel * 8;
    if constexpr (MASKED) {  // branchless causal mask -> v_cndmask
      s[0][r] = (j0 + lane16 > rowAbs) ? -1e9f : s[0][r];
      s[1][r] = (j0 + 16 + lane16 > rowAbs) ? -1e9f : s[1][r];
    }
    const float t = row_max16(fmaxf(s[0][r], s[1][r]));
    const float mnew = fmaxf(mrow[r], t);
    const float corr = __builtin_amdgcn_exp2f(mrow[r] - mnew);
    const float p0 = __builtin_amdgcn_exp2f(s[0][r] - mnew);
    const float p1 = __builtin_amdgcn_exp2f(s[1][r] - mnew);
    const float rs = row_sum16(p0 + p1);
    lrow[r] = lrow[r] * corr + rs;
    mrow[r] = mnew;
#pragma unroll
    for (int nt = 0; nt < 4; ++nt) oacc[nt][r] *= corr;
    myP[(r + sel * 8) * PS + lane16] = (half_t)p0;
    myP[(r + sel * 8) * PS + 16 + lane16] = (half_t)p1;
  }
  // reload P in A layout (same-wave LDS RAW; compiler inserts s_wait_dscnt)
  const half_t* pp = myP + lane16 * PS + sel * 8;
  const v16h pF = combine(*(const v8h*)pp, *(const v8h*)(pp + 16));
#pragma unroll
  for (int nt = 0; nt < 4; ++nt) {
    const half_t* vp = vT + vbase + (size_t)(nt * 16 + lane16) * 2048 + j0 + sel * 8;
    const v16h vF = combine(*(const v8h*)vp, *(const v8h*)(vp + 16));
    oacc[nt] = wmma(pF, vF, oacc[nt]);
  }
}

__global__ __launch_bounds__(256) void attention(
    const half_t* __restrict__ q, const half_t* __restrict__ k,
    const half_t* __restrict__ vT, half_t* __restrict__ o) {
  constexpr int PS = 40;  // padded P row stride (halves)
  __shared__ half_t sP[8][16 * PS];
  const int tid = threadIdx.x, lane = tid & 31, wid = tid >> 5;
  const int lane16 = lane & 15, sel = lane >> 4;
  const int widG = blockIdx.x * 8 + wid;
  const int qt = widG & 127, h = (widG >> 7) & 15, b = widG >> 11;
  const int m0 = qt * 16;
  const int bT = b * 2048, hoff = h * 64;
  const size_t vbase = (((size_t)b * 16 + h) * 64) * 2048;
  half_t* myP = sP[wid];

  // Q fragments (A layout), hd=64 split into two K=32 chunks
  v16h a0, a1;
  {
    const size_t qrow = ((size_t)(bT + m0 + lane16)) * 1024 + hoff;
    const half_t* p0 = q + qrow + sel * 8;
    a0 = combine(*(const v8h*)p0, *(const v8h*)(p0 + 16));
    const half_t* p1 = q + qrow + 32 + sel * 8;
    a1 = combine(*(const v8h*)p1, *(const v8h*)(p1 + 16));
  }

  float mrow[8], lrow[8];
  v8f oacc[4];
  const v8f vzero = {};
#pragma unroll
  for (int r = 0; r < 8; ++r) { mrow[r] = -1e30f; lrow[r] = 0.f; }
#pragma unroll
  for (int nt = 0; nt < 4; ++nt) oacc[nt] = vzero;

  const int nkt = (m0 + 47) >> 5;  // tiles covering keys 0 .. m0+15; last is diagonal
  for (int jt = 0; jt < nkt - 1; ++jt)
    attn_tile<false>(jt * 32, m0, bT, hoff, vbase, lane16, sel, k, vT, myP,
                     a0, a1, mrow, lrow, oacc);
  attn_tile<true>((nkt - 1) * 32, m0, bT, hoff, vbase, lane16, sel, k, vT, myP,
                  a0, a1, mrow, lrow, oacc);

#pragma unroll
  for (int r = 0; r < 8; ++r) {
    const float inv = 1.0f / lrow[r];
    const size_t orow = ((size_t)(bT + m0 + r + sel * 8)) * 1024 + hoff;
#pragma unroll
    for (int nt = 0; nt < 4; ++nt)
      o[orow + nt * 16 + lane16] = (half_t)(oacc[nt][r] * inv);
  }
}

// ---------------------------------------------------------------------------
extern "C" void kernel_launch(void* const* d_in, const int* in_sizes, int n_in,
                              void* d_out, int out_size, void* d_ws, size_t ws_size,
                              hipStream_t stream) {
  (void)in_sizes; (void)n_in; (void)out_size; (void)ws_size;
  const float* x    = (const float*)d_in[0];
  const float* x0   = (const float*)d_in[1];
  const float* lr   = (const float*)d_in[2];
  const float* lx0  = (const float*)d_in[3];
  const float* cosT = (const float*)d_in[4];
  const float* sinT = (const float*)d_in[5];
  const float* qW   = (const float*)d_in[6];
  const float* kW   = (const float*)d_in[7];
  const float* vW   = (const float*)d_in[8];
  const float* pW   = (const float*)d_in[9];
  const float* fcW  = (const float*)d_in[10];
  const float* mlW  = (const float*)d_in[11];

  constexpr size_t M = 4096, C = 1024, F = 4096;
  char* w = (char*)d_ws;
  auto alloc = [&](size_t bytes) -> void* {
    void* p = (void*)w;
    w += (bytes + 255) & ~(size_t)255;
    return p;
  };
  float*  scaled = (float*) alloc(M * C * 4);   // f32 residual stream
  half_t* pre16  = (half_t*)alloc(M * C * 2);   // rmsnorm output (reused for MLP)
  half_t* WqT    = (half_t*)alloc(C * C * 2);
  half_t* WkT    = (half_t*)alloc(C * C * 2);
  half_t* WvT    = (half_t*)alloc(C * C * 2);
  half_t* WpT    = (half_t*)alloc(C * C * 2);
  half_t* WfcT   = (half_t*)alloc(C * F * 2);
  half_t* WmlT   = (half_t*)alloc(F * C * 2);
  half_t* q16    = (half_t*)alloc(M * C * 2);   // rotary applied in place
  half_t* k16    = (half_t*)alloc(M * C * 2);
  half_t* vT16   = (half_t*)alloc(M * C * 2);   // [b][h][d][t]
  half_t* attn16 = (half_t*)alloc(M * C * 2);
  half_t* h16    = (half_t*)alloc(M * F * 2);   // relu^2 MLP hidden

  // 1. weight convert/transpose
  convT<<<2048, 256, 0, stream>>>(qW,  WqT,  1024, 1024);
  convT<<<2048, 256, 0, stream>>>(kW,  WkT,  1024, 1024);
  convT<<<2048, 256, 0, stream>>>(vW,  WvT,  1024, 1024);
  convT<<<2048, 256, 0, stream>>>(pW,  WpT,  1024, 1024);
  convT<<<2048, 256, 0, stream>>>(fcW, WfcT, 1024, 4096);
  convT<<<2048, 256, 0, stream>>>(mlW, WmlT, 4096, 1024);

  // 2. scaled = x*lr + x0*lx0 ; pre = rmsnorm(scaled)
  resid_rmsnorm<<<4096, 256, 0, stream>>>(x, x0, lr, lx0, scaled, pre16, 1);

  // 3. QKV projections (V written pre-transposed for attention)
  gemm256<0><<<dim3(4, 32), 256, 0, stream>>>(pre16, WqT, q16,  nullptr, 4096, 1024, 1024);
  gemm256<0><<<dim3(4, 32), 256, 0, stream>>>(pre16, WkT, k16,  nullptr, 4096, 1024, 1024);
  gemm256<3><<<dim3(4, 32), 256, 0, stream>>>(pre16, WvT, vT16, nullptr, 4096, 1024, 1024);

  // 4. rotary + head rmsnorm (q carries log2e/sqrt(hd))
  rotary_rms<<<8192, 256, 0, stream>>>(q16, k16, cosT, sinT);

  // 5. causal flash attention
  attention<<<512, 256, 0, stream>>>(q16, k16, vT16, attn16);

  // 6. scaled += attn @ projW  (in-place residual add)
  gemm256<1><<<dim3(4, 32), 256, 0, stream>>>(attn16, WpT, scaled, scaled, 4096, 1024, 1024);

  // 7. pre2 = rmsnorm(scaled)
  resid_rmsnorm<<<4096, 256, 0, stream>>>(scaled, nullptr, nullptr, nullptr,
                                          nullptr, pre16, 0);

  // 8. h = relu(pre2 @ fcW)^2
  gemm256<2><<<dim3(16, 32), 256, 0, stream>>>(pre16, WfcT, h16, nullptr, 4096, 4096, 1024);

  // 9. out = scaled + h @ mlpProjW
  gemm256<1><<<dim3(4, 32), 256, 0, stream>>>(h16, WmlT, d_out, scaled, 4096, 1024, 4096);
}
